// SerialAttn_4655744549553
// MI455X (gfx1250) — matrix operand order repaired and checked by reading, not compiled, not run on hardware
//
#include <hip/hip_runtime.h>

// Problem constants (match reference)
#define BQ   8
#define SEQ  4096
#define CH   768
#define C3   2304
#define HN   12
#define DH   64
#define PP   256
#define BH   (BQ * HN)
#define MROWS (BQ * SEQ)
#define EPSN 1e-12f

typedef __attribute__((ext_vector_type(16))) __bf16 v16bf;
typedef __attribute__((ext_vector_type(8)))  float  v8f;

__device__ __forceinline__ unsigned short f2bf(float f) {
  union { float f; unsigned u; } v; v.f = f;
  unsigned u = v.u;
  return (unsigned short)((u + 0x7fffu + ((u >> 16) & 1u)) >> 16);
}

// ---------------- flat fp32 -> bf16 convert (grid-stride) ----------------
__global__ void k_convert_bf16(const float* __restrict__ src,
                               unsigned short* __restrict__ dst, long n) {
  long i = (long)blockIdx.x * blockDim.x + threadIdx.x;
  long stride = (long)gridDim.x * blockDim.x;
  for (; i < n; i += stride) dst[i] = f2bf(src[i]);
}

// ---------------- batched bf16 WMMA GEMM: C(MxN) = A(MxK) @ B(KxN) [+bias] ----
// 128x256x32 block tile, 256 threads = 8 wave32 (2x4), 64x64 per wave,
// double-buffered LDS, branch-free clamped b128 global loads,
// B tile stored K-pair-packed so fragments are plain b32 loads (no shifts).
#define BM 128
#define BN 256
#define BKT 32
#define APAD 20   // uints per A row (16 data + 4 pad) -> 80B, 16B aligned
#define BPAD 260  // uints per packed-B row (256 data + 4 pad) -> 1040B, 16B aligned

__global__ __launch_bounds__(256) void k_gemm_bf16(
    const unsigned short* __restrict__ Ag, const unsigned short* __restrict__ Bg,
    float* __restrict__ Cg, const float* __restrict__ bias,
    int M, int Nn, int K, long sA, long sB, long sC)
{
  __shared__ unsigned aT[2][BM][APAD];         // A: row-major K-pairs
  __shared__ unsigned bTp[2][BKT / 2][BPAD];   // B: [kpair][n] packed (lo|hi<<16)

  const int tid  = threadIdx.x;
  const int lane = tid & 31;
  const int wave = tid >> 5;
  const int waveRow = wave >> 2;   // 0..1 -> 64-row strip
  const int waveCol = wave & 3;    // 0..3 -> 64-col strip

  const int m0 = blockIdx.y * BM;
  const int n0 = blockIdx.x * BN;
  const long zb = blockIdx.z;
  const unsigned* A32 = (const unsigned*)(Ag + zb * sA);
  const unsigned* B32 = (const unsigned*)(Bg + zb * sB);
  float* C = Cg + zb * sC;

  const int Kp = K >> 1;   // uints per A row
  const int Np = Nn >> 1;  // uints per B row

  // --- A staging: 2 threads per row, 8 consecutive uints (32B) each ---
  const int arow  = tid >> 1;
  const int akk   = (tid & 1) * 8;
  const int agRow = m0 + arow;
  const int acRow = agRow < M ? agRow : (M - 1);
  const unsigned amask = (agRow < M) ? 0xffffffffu : 0u;
  // --- B staging: thread owns K-pair row pr (0..15) and 16 columns ---
  const int pr = tid >> 4;           // k-pair row in tile
  const int bnc = (tid & 15) * 8;    // source uint offset (16 columns)
  const bool bok = (n0 + 2 * bnc) < Nn;     // Nn % 16 == 0 -> all-or-nothing
  const int bncc = bok ? bnc : 0;
  const unsigned bmask = bok ? 0xffffffffu : 0u;

  uint4 ra0, ra1, rg0, rg1, rh0, rh1;

  auto gload = [&](int k0) {
    const uint4* asrc = (const uint4*)(A32 + (long)acRow * Kp + (k0 >> 1) + akk);
    ra0 = asrc[0]; ra1 = asrc[1];
    const uint4* bsrcL = (const uint4*)(B32 + (long)(k0 + 2 * pr) * Np + (n0 >> 1) + bncc);
    const uint4* bsrcH = (const uint4*)(B32 + (long)(k0 + 2 * pr + 1) * Np + (n0 >> 1) + bncc);
    rg0 = bsrcL[0]; rg1 = bsrcL[1];
    rh0 = bsrcH[0]; rh1 = bsrcH[1];
  };
  auto stolds = [&](int buf) {
    uint4 a0 = ra0, a1 = ra1;
    a0.x &= amask; a0.y &= amask; a0.z &= amask; a0.w &= amask;
    a1.x &= amask; a1.y &= amask; a1.z &= amask; a1.w &= amask;
    *(uint4*)&aT[buf][arow][akk]     = a0;
    *(uint4*)&aT[buf][arow][akk + 4] = a1;
    // pack B pairs: o[2i] = lo16(g)|hi<<16 ; o[2i+1] = g>>16 | (h & hi16)
    unsigned g[8] = { rg0.x, rg0.y, rg0.z, rg0.w, rg1.x, rg1.y, rg1.z, rg1.w };
    unsigned h[8] = { rh0.x, rh0.y, rh0.z, rh0.w, rh1.x, rh1.y, rh1.z, rh1.w };
    unsigned o[16];
    #pragma unroll
    for (int i = 0; i < 8; ++i) {
      unsigned gi = g[i] & bmask, hi = h[i] & bmask;
      o[2 * i]     = (gi & 0x0000ffffu) | (hi << 16);
      o[2 * i + 1] = (gi >> 16)         | (hi & 0xffff0000u);
    }
    #pragma unroll
    for (int q = 0; q < 4; ++q) {
      uint4 t = { o[4 * q], o[4 * q + 1], o[4 * q + 2], o[4 * q + 3] };
      *(uint4*)&bTp[buf][pr][2 * bnc + 4 * q] = t;
    }
  };

  v8f acc[4][4];
  #pragma unroll
  for (int i = 0; i < 4; ++i)
    #pragma unroll
    for (int j = 0; j < 4; ++j) { v8f z = {}; acc[i][j] = z; }

  const int lm  = lane & 15;
  const int sel = lane >> 4;

  gload(0);
  stolds(0);
  int cur = 0;

  for (int k0 = 0; k0 < K; k0 += BKT) {
    __syncthreads();
    const bool more = (k0 + BKT) < K;
    if (more) {
      gload(k0 + BKT);
      if (k0 + 2 * BKT < K) {   // warm L2 two tiles ahead (global_prefetch_b8)
        __builtin_prefetch(A32 + (long)acRow * Kp + ((k0 + 2 * BKT) >> 1) + akk, 0, 1);
        __builtin_prefetch(B32 + (long)(k0 + 2 * BKT + 2 * pr) * Np + (n0 >> 1) + bncc, 0, 1);
      }
    }

    union Frag { v16bf v; unsigned u[8]; };
    Frag aF[4];
    #pragma unroll
    for (int tm = 0; tm < 4; ++tm) {
      int row = waveRow * 64 + tm * 16 + lm;
      #pragma unroll
      for (int j = 0; j < 4; ++j) {
        aF[tm].u[j]     = aT[cur][row][sel * 4 + j];       // K = sel*8 + 2j (+1)
        aF[tm].u[4 + j] = aT[cur][row][8 + sel * 4 + j];   // K = 16 + sel*8 + 2j
      }
    }
    #pragma unroll
    for (int tn = 0; tn < 4; ++tn) {
      int col = waveCol * 64 + tn * 16 + lm;
      Frag bF;
      #pragma unroll
      for (int j = 0; j < 8; ++j)
        bF.u[j] = bTp[cur][(j >> 2) * 8 + sel * 4 + (j & 3)][col];   // K = (j>>2)*16 + sel*8 + 2(j&3) (+1), packed
      #pragma unroll
      for (int tm = 0; tm < 4; ++tm)
        acc[tm][tn] = __builtin_amdgcn_wmma_f32_16x16x32_bf16(
            false, aF[tm].v, false, bF.v, (short)0, acc[tm][tn], false, false);
    }

    if (more) stolds(cur ^ 1);
    cur ^= 1;
  }

  #pragma unroll
  for (int tm = 0; tm < 4; ++tm)
    #pragma unroll
    for (int tn = 0; tn < 4; ++tn)
      #pragma unroll
      for (int r = 0; r < 8; ++r) {
        int m = m0 + waveRow * 64 + tm * 16 + sel * 8 + r;
        int n = n0 + waveCol * 64 + tn * 16 + lm;
        if (m < M && n < Nn) {
          float v = acc[tm][tn][r];
          if (bias) v += bias[n];
          C[(long)m * Nn + n] = v;
        }
      }
}

// ---------------- per-(b,c) L2 norm over N (stride C3 column) ----------------
__global__ void k_colnorm(const float* __restrict__ qkv,
                          float* __restrict__ norms, int section) {
  int col = blockIdx.x;            // 0 .. BQ*CH-1
  int b = col / CH;
  int c = col % CH;
  const float* p = qkv + (long)b * SEQ * C3 + section * CH + c;
  float s = 0.f;
  for (int n = threadIdx.x; n < SEQ; n += blockDim.x) {
    float v = p[(long)n * C3];
    s += v * v;
  }
  __shared__ float red[256];
  red[threadIdx.x] = s;
  __syncthreads();
  for (int offi = 128; offi > 0; offi >>= 1) {
    if (threadIdx.x < offi) red[threadIdx.x] += red[threadIdx.x + offi];
    __syncthreads();
  }
  if (threadIdx.x == 0) norms[col] = sqrtf(red[0]);
}

// ---------------- gather head slice of qkv into per-(b,h) bf16 matrix --------
__global__ void k_slice(const float* __restrict__ qkv,
                        unsigned short* __restrict__ dst,
                        const float* __restrict__ norms,
                        int section, int transposed) {
  long idx = (long)blockIdx.x * blockDim.x + threadIdx.x;
  long total = (long)BH * SEQ * DH;
  if (idx >= total) return;
  int d = (int)(idx % DH);
  long t = idx / DH;
  int n = (int)(t % SEQ);
  int bh = (int)(t / SEQ);
  int h = bh % HN;
  int b = bh / HN;
  float v = qkv[((long)b * SEQ + n) * C3 + section * CH + h * DH + d];
  if (norms) {
    float nm = norms[b * CH + h * DH + d];
    v = v / fmaxf(nm, EPSN);
  }
  long base = (long)bh * (SEQ * DH);
  long off = transposed ? ((long)d * SEQ + n) : ((long)n * DH + d);
  dst[base + off] = f2bf(v);
}

// ---------------- row softmax with per-head temperature ----------------------
__global__ void k_softmax(float* __restrict__ data, int row_len, int rows_per_h,
                          const float* __restrict__ scale) {
  long row = blockIdx.x;
  int h = (int)((row / rows_per_h) % HN);
  float sc = scale[h];
  float* p = data + row * row_len;
  int tid = threadIdx.x;
  __shared__ float red[256];

  float m = -3.4e38f;
  for (int i = tid; i < row_len; i += blockDim.x) m = fmaxf(m, p[i] * sc);
  red[tid] = m; __syncthreads();
  for (int o = 128; o > 0; o >>= 1) {
    if (tid < o) red[tid] = fmaxf(red[tid], red[tid + o]);
    __syncthreads();
  }
  m = red[0]; __syncthreads();

  float s = 0.f;
  for (int i = tid; i < row_len; i += blockDim.x) {
    float e = __expf(p[i] * sc - m);
    p[i] = e; s += e;
  }
  red[tid] = s; __syncthreads();
  for (int o = 128; o > 0; o >>= 1) {
    if (tid < o) red[tid] += red[tid + o];
    __syncthreads();
  }
  s = red[0];
  float inv = 1.f / s;
  for (int i = tid; i < row_len; i += blockDim.x) p[i] *= inv;
}

// ---------------- (bh,D,P) f32 -> (bh,P,D) bf16 transpose --------------------
__global__ void k_transpose_pd(const float* __restrict__ src,
                               unsigned short* __restrict__ dst) {
  long idx = (long)blockIdx.x * blockDim.x + threadIdx.x;
  long total = (long)BH * DH * PP;
  if (idx >= total) return;
  int p = (int)(idx % PP);
  long t = idx / PP;
  int d = (int)(t % DH);
  int bh = (int)(t / DH);
  dst[(long)bh * PP * DH + (long)p * DH + d] = f2bf(src[idx]);
}

// --- faithful (B,H,N,D).transpose(0,3,1,2).reshape(B,N,C): flat=d*H*N+h*N+n --
__global__ void k_scramble_bf16(const float* __restrict__ src,
                                unsigned short* __restrict__ dst) {
  long idx = (long)blockIdx.x * blockDim.x + threadIdx.x;
  long total = (long)BQ * HN * SEQ * DH;
  if (idx >= total) return;
  int d = (int)(idx % DH);
  long t = idx / DH;
  int n = (int)(t % SEQ);
  int bh = (int)(t / SEQ);
  int h = bh % HN;
  int b = bh / HN;
  long flat = (long)d * (HN * SEQ) + (long)h * SEQ + n;
  dst[(long)b * (SEQ * CH) + flat] = f2bf(src[idx]);
}

__global__ void k_scramble_f32(const float* __restrict__ src,
                               float* __restrict__ dst) {
  long idx = (long)blockIdx.x * blockDim.x + threadIdx.x;
  long total = (long)BQ * HN * SEQ * DH;
  if (idx >= total) return;
  int d = (int)(idx % DH);
  long t = idx / DH;
  int n = (int)(t % SEQ);
  int bh = (int)(t / SEQ);
  int h = bh % HN;
  int b = bh / HN;
  long flat = (long)d * (HN * SEQ) + (long)h * SEQ + n;
  dst[(long)b * (SEQ * CH) + flat] = src[idx];
}

// ============================================================================
extern "C" void kernel_launch(void* const* d_in, const int* in_sizes, int n_in,
                              void* d_out, int out_size, void* d_ws, size_t ws_size,
                              hipStream_t stream) {
  (void)in_sizes; (void)n_in; (void)out_size; (void)ws_size;
  const float* x     = (const float*)d_in[0];
  const float* Wqkv2 = (const float*)d_in[1];
  const float* Wqkv  = (const float*)d_in[2];
  const float* Wproj = (const float*)d_in[3];
  const float* bproj = (const float*)d_in[4];
  const float* temp  = (const float*)d_in[5];
  const float* temp2 = (const float*)d_in[6];
  float* out = (float*)d_out;

  char* ws = (char*)d_ws;
  size_t off = 0;
  auto alloc = [&](size_t bytes) -> void* {
    void* p = ws + off;
    off += (bytes + 255) & ~(size_t)255;
    return p;
  };

  unsigned short* xb   = (unsigned short*)alloc((size_t)MROWS * CH * 2);
  unsigned short* wb2  = (unsigned short*)alloc((size_t)CH * C3 * 2);
  unsigned short* wb   = (unsigned short*)alloc((size_t)CH * C3 * 2);
  unsigned short* wpb  = (unsigned short*)alloc((size_t)SEQ * PP * 2);
  float*          qkv  = (float*)alloc((size_t)MROWS * C3 * 4);        // reused for both stages
  float*          nrm  = (float*)alloc((size_t)BQ * CH * 4);
  unsigned short* bhA  = (unsigned short*)alloc((size_t)BH * SEQ * DH * 2);
  unsigned short* bhB  = (unsigned short*)alloc((size_t)BH * SEQ * DH * 2);
  float*          kpf  = (float*)alloc((size_t)BH * DH * PP * 4);
  float*          vpf  = (float*)alloc((size_t)BH * DH * PP * 4);
  unsigned short* kpb  = (unsigned short*)alloc((size_t)BH * DH * PP * 2);
  unsigned short* vptb = (unsigned short*)alloc((size_t)BH * PP * DH * 2);
  float*          attn = (float*)alloc((size_t)BH * SEQ * PP * 4);
  unsigned short* attnb= (unsigned short*)alloc((size_t)BH * SEQ * PP * 2);
  float*          xsaf = (float*)alloc((size_t)BH * SEQ * DH * 4);     // also reused for x_ca
  unsigned short* xsb  = (unsigned short*)alloc((size_t)MROWS * CH * 2);
  float*          acaf = (float*)alloc((size_t)BH * DH * DH * 4);
  unsigned short* acab = (unsigned short*)alloc((size_t)BH * DH * DH * 2);

  const long sliceTot = (long)BH * SEQ * DH;
  const int sliceBlk = (int)((sliceTot + 255) / 256);
  auto cvt = [&](const float* s, unsigned short* d, long n) {
    int blk = (int)(((n + 255) / 256) > 262144 ? 262144 : ((n + 255) / 256));
    k_convert_bf16<<<blk, 256, 0, stream>>>(s, d, n);
  };

  // ---- bf16 weight / input prep ----
  cvt(x,     xb,  (long)MROWS * CH);
  cvt(Wqkv2, wb2, (long)CH * C3);
  cvt(Wqkv,  wb,  (long)CH * C3);
  cvt(Wproj, wpb, (long)SEQ * PP);

  // ---- GEMM1: qkv2 = x @ Wqkv2 ----
  k_gemm_bf16<<<dim3((C3 + BN - 1) / BN, MROWS / BM, 1), 256, 0, stream>>>(
      xb, wb2, qkv, nullptr, MROWS, C3, CH, 0, 0, 0);

  // ---- k_proj = k2^T @ Wproj + bproj  (batched over BH, M=64,N=256,K=4096) ----
  k_slice<<<sliceBlk, 256, 0, stream>>>(qkv, bhA, nullptr, 1, 1);   // (D x N) per bh
  k_gemm_bf16<<<dim3(1, 1, BH), 256, 0, stream>>>(
      bhA, wpb, kpf, bproj, DH, PP, SEQ, (long)SEQ * DH, 0, (long)DH * PP);

  // ---- v_proj = v2^T @ Wproj + bproj ----
  k_slice<<<sliceBlk, 256, 0, stream>>>(qkv, bhA, nullptr, 2, 1);
  k_gemm_bf16<<<dim3(1, 1, BH), 256, 0, stream>>>(
      bhA, wpb, vpf, bproj, DH, PP, SEQ, (long)SEQ * DH, 0, (long)DH * PP);

  // ---- q2n: L2-normalize q2 over N, gather (N x D) per bh ----
  k_colnorm<<<BQ * CH, 256, 0, stream>>>(qkv, nrm, 0);
  k_slice<<<sliceBlk, 256, 0, stream>>>(qkv, bhA, nrm, 0, 0);

  // ---- attn_sa = q2n @ k_proj  (M=4096,N=256,K=64) ----
  cvt(kpf, kpb, (long)BH * DH * PP);
  k_gemm_bf16<<<dim3(1, SEQ / BM, BH), 256, 0, stream>>>(
      bhA, kpb, attn, nullptr, SEQ, PP, DH, (long)SEQ * DH, (long)DH * PP, (long)SEQ * PP);

  // ---- softmax(attn * temp2) over P ----
  k_softmax<<<BH * SEQ, 256, 0, stream>>>(attn, PP, SEQ, temp2);
  cvt(attn, attnb, (long)BH * SEQ * PP);

  // ---- x_sa = attn @ v_proj^T  (M=4096,N=64,K=256) ----
  {
    long n = (long)BH * DH * PP;
    k_transpose_pd<<<(int)((n + 255) / 256), 256, 0, stream>>>(vpf, vptb);
  }
  k_gemm_bf16<<<dim3(1, SEQ / BM, BH), 256, 0, stream>>>(
      attnb, vptb, xsaf, nullptr, SEQ, DH, PP, (long)SEQ * PP, (long)PP * DH, (long)SEQ * DH);

  // ---- faithful permute+reshape to (B,N,C), bf16 for GEMM2 ----
  k_scramble_bf16<<<sliceBlk, 256, 0, stream>>>(xsaf, xsb);

  // ---- GEMM2: qkv = x_sa @ Wqkv (overwrites qkv buffer) ----
  k_gemm_bf16<<<dim3((C3 + BN - 1) / BN, MROWS / BM, 1), 256, 0, stream>>>(
      xsb, wb, qkv, nullptr, MROWS, C3, CH, 0, 0, 0);

  // ---- channel attention: qn^T @ kn ----
  k_colnorm<<<BQ * CH, 256, 0, stream>>>(qkv, nrm, 0);
  k_slice<<<sliceBlk, 256, 0, stream>>>(qkv, bhA, nrm, 0, 1);   // qn^T (D x N)
  k_colnorm<<<BQ * CH, 256, 0, stream>>>(qkv, nrm, 1);
  k_slice<<<sliceBlk, 256, 0, stream>>>(qkv, bhB, nrm, 1, 0);   // kn (N x D)
  k_gemm_bf16<<<dim3(1, 1, BH), 256, 0, stream>>>(
      bhA, bhB, acaf, nullptr, DH, DH, SEQ, (long)SEQ * DH, (long)SEQ * DH, (long)DH * DH);

  // ---- softmax(attn_ca * temp) over last D ----
  k_softmax<<<BH * DH, 256, 0, stream>>>(acaf, DH, DH, temp);
  cvt(acaf, acab, (long)BH * DH * DH);

  // ---- x_ca = v @ attn_ca  (M=4096,N=64,K=64) ----
  k_slice<<<sliceBlk, 256, 0, stream>>>(qkv, bhA, nullptr, 2, 0);  // v (N x D)
  k_gemm_bf16<<<dim3(1, SEQ / BM, BH), 256, 0, stream>>>(
      bhA, acab, xsaf, nullptr, SEQ, DH, DH, (long)SEQ * DH, (long)DH * DH, (long)SEQ * DH);

  // ---- final faithful permute+reshape -> out (fp32) ----
  k_scramble_f32<<<sliceBlk, 256, 0, stream>>>(xsaf, out);
}
